// GAT_61495341744418
// MI455X (gfx1250) — matrix operand order repaired and checked by reading, not compiled, not run on hardware
//
#include <hip/hip_runtime.h>
#include <hip/hip_bf16.h>

typedef __attribute__((ext_vector_type(16))) _Float16 v16h;
typedef __attribute__((ext_vector_type(8)))  float    v8f;

#define NN 20000
#define EE 100000
#define EA (EE + NN)
#define DD 64
#define HH 16
#define CC 64
#define HC (HH * CC)

#define ENC_NEGINF 0x007FFFFFu

// Monotone float->uint mapping so atomicMax(uint) implements float max
__device__ __forceinline__ unsigned encf(float f) {
    unsigned u = __float_as_uint(f);
    return (u & 0x80000000u) ? ~u : (u | 0x80000000u);
}
__device__ __forceinline__ float decf(unsigned u) {
    unsigned v = (u & 0x80000000u) ? (u & 0x7FFFFFFFu) : ~u;
    return __uint_as_float(v);
}

// ---- WMMA fragment-order indexing (ISA 7.12.2 layouts, K = 64 as 2 ksteps of 32) ----
// A matrix [M,64] f16: frag = [tileM][kstep][lane][elem16]
//   lane = hi*16 + (row&15);  elem i: K_local = ((i&8)?16:0) + hi*8 + (i&7)
__device__ __forceinline__ int fragA_idx(int row, int k) {
    int tm = row >> 4, lr = row & 15;
    int s  = k >> 5,  kl = k & 31;
    int hi = (kl >> 3) & 1;
    int i  = (kl & 7) | ((kl & 16) ? 8 : 0);
    return ((tm * 2 + s) * 32 + hi * 16 + lr) * 16 + i;
}
// B matrix [64,Ncols] f16: frag = [tileN][kstep][lane][elem16]
//   lane = hi*16 + (col&15);  elem i: K_local = ((i&8)?16:0) + hi*8 + (i&7)
__device__ __forceinline__ int fragB_idx(int k, int col) {
    int tn = col >> 4, lr = col & 15;
    int s  = k >> 5,  kl = k & 31;
    int hi = (kl >> 3) & 1;
    int i  = (kl & 7) | ((kl & 16) ? 8 : 0);
    return ((tn * 2 + s) * 32 + hi * 16 + lr) * 16 + i;
}

// ---------------- self-loop mean edge_attr ----------------
__global__ void k_loop_init(float* loop_sum, float* cnt) {
    int t = blockIdx.x * blockDim.x + threadIdx.x;
    if (t < NN * DD) loop_sum[t] = 0.f;
    if (t < NN) cnt[t] = 0.f;
}

__global__ void k_degree(const int* __restrict__ ei, const float* __restrict__ ea,
                         float* loop_sum, float* cnt) {
    int t = blockIdx.x * blockDim.x + threadIdx.x;
    if (t >= EE * DD) return;
    int e = t / DD, d = t % DD;
    int dn = ei[EE + e];
    atomicAdd(loop_sum + dn * DD + d, ea[t]);
    if (d == 0) atomicAdd(cnt + dn, 1.0f);
}

__global__ void k_loop_mean(float* loop_sum, const float* __restrict__ cnt) {
    int t = blockIdx.x * blockDim.x + threadIdx.x;
    if (t >= NN * DD) return;
    loop_sum[t] = loop_sum[t] / fmaxf(cnt[t / DD], 1.0f);
}

// ---------------- pack fp32 matrices into f16 WMMA fragment order ----------------
__global__ void k_pack_a(const float* __restrict__ src, _Float16* dst, int total) {
    int t = blockIdx.x * blockDim.x + threadIdx.x;
    if (t >= total) return;
    int row = t / DD, k = t % DD;
    dst[fragA_idx(row, k)] = (_Float16)src[t];
}

// concatenated edge features [EA,64] -> A-fragment order
__global__ void k_pack_ea(const float* __restrict__ edge_attr,
                          const float* __restrict__ loop_ea, _Float16* dst) {
    int t = blockIdx.x * blockDim.x + threadIdx.x;
    if (t >= EA * DD) return;
    int e = t / DD, d = t % DD;
    float v = (e < EE) ? edge_attr[t] : loop_ea[(e - EE) * DD + d];
    dst[fragA_idx(e, d)] = (_Float16)v;
}

__global__ void k_pack_b(const float* __restrict__ W, _Float16* dst, int Ncols) {
    int t = blockIdx.x * blockDim.x + threadIdx.x;
    if (t >= DD * Ncols) return;
    int k = t / Ncols, col = t % Ncols;
    dst[fragB_idx(k, col)] = (_Float16)W[t];
}

// w_e[d,h] = sum_c We[d, h*C + c] * a_e[h,c]  (folds lin_edge + att_edge)
__global__ void k_fold_we(const float* __restrict__ We, const float* __restrict__ ae,
                          float* we_f32) {
    int t = blockIdx.x * blockDim.x + threadIdx.x;
    if (t >= DD * HH) return;
    int d = t / HH, h = t % HH;
    float s = 0.f;
    for (int c = 0; c < CC; ++c) s += We[d * HC + h * CC + c] * ae[h * CC + c];
    we_f32[d * HH + h] = s;   // row-major [64,16], packed by k_pack_b
}

// ---------------- WMMA GEMM: C[M,Ncols] = A[M,64] * B[64,Ncols] ----------------
// Operands pre-swizzled to fragment order: one v16h (32B) load per lane per fragment.
__global__ void k_wmma_gemm(const v16h* __restrict__ Af, const v16h* __restrict__ Bf,
                            float* __restrict__ Cm, int M, int Ncols) {
    int wid  = threadIdx.x >> 5;
    int lane = threadIdx.x & 31;
    int gw = blockIdx.x * (blockDim.x >> 5) + wid;
    int tilesN = Ncols >> 4;
    if (gw >= (M >> 4) * tilesN) return;  // wave-uniform: EXEC all-ones around WMMA
    int tm = gw / tilesN, tn = gw % tilesN;

    v8f acc = {};
#pragma unroll
    for (int s = 0; s < 2; ++s) {
        v16h a = Af[(tm * 2 + s) * 32 + lane];
        v16h b = Bf[(tn * 2 + s) * 32 + lane];
        acc = __builtin_amdgcn_wmma_f32_16x16x32_f16(
            /*neg_a=*/false, a, /*neg_b=*/false, b,
            /*c_mod=*/(short)0, acc, /*reuse_a=*/false, /*reuse_b=*/false);
    }
    // C/D 16x16 f32: VGPR r -> row r + 8*hi, col = lane&15
    int hi = lane >> 4, lr = lane & 15;
#pragma unroll
    for (int r = 0; r < 8; ++r)
        Cm[(tm * 16 + r + 8 * hi) * Ncols + tn * 16 + lr] = acc[r];
}

// ---------------- attention logits per node ----------------
__global__ void k_atts(const float* __restrict__ xp, const float* __restrict__ asrc,
                       const float* __restrict__ adst, float* al_s, float* al_d) {
    int t = blockIdx.x * blockDim.x + threadIdx.x;
    if (t >= NN * HH) return;
    int n = t / HH, h = t % HH;
    const float* xr = xp + n * HC + h * CC;
    float s = 0.f, d = 0.f;
    for (int c = 0; c < CC; ++c) {
        float v = xr[c];
        s += v * asrc[h * CC + c];
        d += v * adst[h * CC + c];
    }
    al_s[t] = s;
    al_d[t] = d;
}

__global__ void k_seg_init(float* acc, float* den, unsigned* menc) {
    int t = blockIdx.x * blockDim.x + threadIdx.x;
    if (t < NN * HC) acc[t] = 0.f;
    if (t < NN * HH) { den[t] = 0.f; menc[t] = ENC_NEGINF; }
}

// ---------------- edge softmax ----------------
__global__ void k_alpha(const int* __restrict__ ei, const float* __restrict__ al_s,
                        const float* __restrict__ al_d, const float* __restrict__ al_e,
                        float* ab, unsigned* menc) {
    int t = blockIdx.x * blockDim.x + threadIdx.x;
    if (t >= EA * HH) return;
    int e = t / HH, h = t % HH;
    int s  = (e < EE) ? ei[e]      : (e - EE);
    int dn = (e < EE) ? ei[EE + e] : (e - EE);
    float a = al_s[s * HH + h] + al_d[dn * HH + h] + al_e[t];
    a = (a > 0.f) ? a : 0.2f * a;            // leaky_relu(0.2)
    ab[t] = a;
    atomicMax(menc + dn * HH + h, encf(a));
}

__global__ void k_exps(const int* __restrict__ ei, float* ab,
                       const unsigned* __restrict__ menc, float* den) {
    int t = blockIdx.x * blockDim.x + threadIdx.x;
    if (t >= EA * HH) return;
    int e = t / HH, h = t % HH;
    int dn = (e < EE) ? ei[EE + e] : (e - EE);
    float ex = __expf(ab[t] - decf(menc[dn * HH + h]));
    ab[t] = ex;
    atomicAdd(den + dn * HH + h, ex);
}

// ---------------- weighted scatter: acc[dst,h,:] += att * xp[src,h,:] ----------------
__global__ void k_aggr(const int* __restrict__ ei, const float* __restrict__ ab,
                       const float* __restrict__ den, const float* __restrict__ xp,
                       float* acc) {
    int t = blockIdx.x * blockDim.x + threadIdx.x;
    if (t >= EA * HH) return;
    int e = t / HH, h = t % HH;
    int s  = (e < EE) ? ei[e]      : (e - EE);
    int dn = (e < EE) ? ei[EE + e] : (e - EE);
    float att = ab[t] / (den[dn * HH + h] + 1e-16f);
    const float* xr = xp + s * HC + h * CC;
    float* ar = acc + dn * HC + h * CC;
    __builtin_prefetch(xr, 0, 0);            // global_prefetch_b8 on gfx1250
#pragma unroll 4
    for (int c = 0; c < CC; ++c) atomicAdd(ar + c, xr[c] * att);
}

// ---------------- head-mean + bias + relu ----------------
// Layer 1 writes f16 directly in A-fragment order (feeds layer-2 WMMA GEMM).
__global__ void k_final(const float* __restrict__ acc, const float* __restrict__ bias,
                        float* out_f32, _Float16* out_f16frag) {
    int t = blockIdx.x * blockDim.x + threadIdx.x;
    if (t >= NN * CC) return;
    int n = t / CC, c = t % CC;
    float s = 0.f;
    for (int h = 0; h < HH; ++h) s += acc[n * HC + h * CC + c];
    float v = fmaxf(s * (1.0f / HH) + bias[c], 0.f);
    if (out_f32) out_f32[t] = v;
    if (out_f16frag) out_f16frag[fragA_idx(n, c)] = (_Float16)v;
}

extern "C" void kernel_launch(void* const* d_in, const int* in_sizes, int n_in,
                              void* d_out, int out_size, void* d_ws, size_t ws_size,
                              hipStream_t stream) {
    (void)in_sizes; (void)n_in; (void)out_size; (void)ws_size;
    const float* x   = (const float*)d_in[0];
    const int*   ei  = (const int*)d_in[1];     // [2,E]: ei[e]=src, ei[E+e]=dst
    const float* ea  = (const float*)d_in[2];
    const float* W1  = (const float*)d_in[3];
    const float* as1 = (const float*)d_in[4];
    const float* ad1 = (const float*)d_in[5];
    const float* We1 = (const float*)d_in[6];
    const float* ae1 = (const float*)d_in[7];
    const float* b1  = (const float*)d_in[8];
    const float* W2  = (const float*)d_in[9];
    const float* as2 = (const float*)d_in[10];
    const float* ad2 = (const float*)d_in[11];
    const float* We2 = (const float*)d_in[12];
    const float* ae2 = (const float*)d_in[13];
    const float* b2  = (const float*)d_in[14];
    float* out = (float*)d_out;

    char* p = (char*)d_ws;
    auto take = [&](size_t bytes) -> char* {
        char* r = p;
        p += (bytes + 255) & ~(size_t)255;
        return r;
    };
    float*    cnt     = (float*)take((size_t)NN * 4);
    float*    loop_ea = (float*)take((size_t)NN * DD * 4);
    _Float16* x16     = (_Float16*)take((size_t)NN * DD * 2);   // A-frag order
    _Float16* h16     = (_Float16*)take((size_t)NN * DD * 2);   // A-frag order
    _Float16* Wf16    = (_Float16*)take((size_t)DD * HC * 2);   // B-frag order
    float*    we_f32  = (float*)take((size_t)DD * HH * 4);
    _Float16* we16    = (_Float16*)take((size_t)DD * HH * 2);   // B-frag order
    _Float16* ea16    = (_Float16*)take((size_t)EA * DD * 2);   // A-frag order
    float*    xp      = (float*)take((size_t)NN * HC * 4);
    float*    al_s    = (float*)take((size_t)NN * HH * 4);
    float*    al_d    = (float*)take((size_t)NN * HH * 4);
    float*    al_e    = (float*)take((size_t)EA * HH * 4);
    float*    ab      = (float*)take((size_t)EA * HH * 4);
    unsigned* menc    = (unsigned*)take((size_t)NN * HH * 4);
    float*    den     = (float*)take((size_t)NN * HH * 4);
    float*    accb    = (float*)take((size_t)NN * HC * 4);

    const int B = 256;
    auto blk = [](long n) { return (int)((n + 255) / 256); };

    // self-loop edge_attr = per-dst mean; pack operands into WMMA fragment order
    k_loop_init<<<blk((long)NN * DD), B, 0, stream>>>(loop_ea, cnt);
    k_degree<<<blk((long)EE * DD), B, 0, stream>>>(ei, ea, loop_ea, cnt);
    k_loop_mean<<<blk((long)NN * DD), B, 0, stream>>>(loop_ea, cnt);
    k_pack_ea<<<blk((long)EA * DD), B, 0, stream>>>(ea, loop_ea, ea16);
    k_pack_a<<<blk((long)NN * DD), B, 0, stream>>>(x, x16, NN * DD);

    for (int layer = 0; layer < 2; ++layer) {
        const float* W    = layer ? W2  : W1;
        const float* as   = layer ? as2 : as1;
        const float* ad   = layer ? ad2 : ad1;
        const float* Wep  = layer ? We2 : We1;
        const float* aew  = layer ? ae2 : ae1;
        const float* bias = layer ? b2  : b1;
        const _Float16* Xin = layer ? h16 : x16;

        k_pack_b<<<blk((long)DD * HC), B, 0, stream>>>(W, Wf16, HC);
        k_fold_we<<<blk((long)DD * HH), B, 0, stream>>>(Wep, aew, we_f32);
        k_pack_b<<<blk((long)DD * HH), B, 0, stream>>>(we_f32, we16, HH);

        // xp = Xin @ W : [20000,1024]   (1250*64 tiles, 8 waves/block)
        {
            int tiles = (NN / 16) * (HC / 16);
            k_wmma_gemm<<<(tiles + 7) / 8, B, 0, stream>>>(
                (const v16h*)Xin, (const v16h*)Wf16, xp, NN, HC);
        }
        // al_e = ea16 @ we16 : [120000,16]   (7500 tiles)
        {
            int tiles = (EA / 16) * (HH / 16);
            k_wmma_gemm<<<(tiles + 7) / 8, B, 0, stream>>>(
                (const v16h*)ea16, (const v16h*)we16, al_e, EA, HH);
        }
        k_atts<<<blk((long)NN * HH), B, 0, stream>>>(xp, as, ad, al_s, al_d);
        k_seg_init<<<blk((long)NN * HC), B, 0, stream>>>(accb, den, menc);
        k_alpha<<<blk((long)EA * HH), B, 0, stream>>>(ei, al_s, al_d, al_e, ab, menc);
        k_exps<<<blk((long)EA * HH), B, 0, stream>>>(ei, ab, menc, den);
        k_aggr<<<blk((long)EA * HH), B, 0, stream>>>(ei, ab, den, xp, accb);
        k_final<<<blk((long)NN * CC), B, 0, stream>>>(accb, bias,
                                                      layer ? out : nullptr,
                                                      layer ? nullptr : h16);
    }
}
